// ComConV_69423851373237
// MI455X (gfx1250) — matrix operand order repaired and checked by reading, not compiled, not run on hardware
//
#include <hip/hip_runtime.h>
#include <hip/hip_bf16.h>

// ---------------- types for WMMA ----------------
typedef __attribute__((ext_vector_type(16))) __bf16 v16bf;
typedef __attribute__((ext_vector_type(8)))  __bf16 v8bf;
typedef __attribute__((ext_vector_type(8)))  float  v8f;

// ---------------- problem constants ----------------
#define ENT_DIM 200
#define KPAD    224            // 200 padded to 7*32 for bf16 WMMA K
#define BATCH   1024
#define NENT    100000
#define NE_PAD  100352         // 196 blocks * 512 cols
#define OUT_CH  32
#define CW      199            // conv output width
#define FCK     6368           // 32*199, exactly 199*32 -> no K padding
#define FCN_PAD 224            // 200 fc outputs padded to 14*16
#define EPS     1e-5f

// workspace layout (bytes, all offsets 256-aligned)
static constexpr size_t ENT_BF_OFF = 0;                                     // 100352*224*2 = 44,957,696
static constexpr size_t FCW_BF_OFF = (size_t)NE_PAD * KPAD * 2;             // +224*6368*2 = 2,852,864
static constexpr size_t Z_OFF      = FCW_BF_OFF + (size_t)FCN_PAD * FCK * 2;// +1024*6368*2 = 13,041,664
static constexpr size_t XFC_OFF    = Z_OFF + (size_t)BATCH * FCK * 2;       // +1024*224*2 = 458,752
// total ~= 61.4 MB of d_ws

// ---------------- f32 -> bf16 with zero padding ----------------
__global__ void cvt_pad_kernel(const float* __restrict__ src, __bf16* __restrict__ dst,
                               int srows, int scols, int drows, int dcols) {
    size_t idx = (size_t)blockIdx.x * blockDim.x + threadIdx.x;
    size_t n = (size_t)drows * dcols;
    if (idx >= n) return;
    int r = (int)(idx / dcols);
    int c = (int)(idx - (size_t)r * dcols);
    float v = (r < srows && c < scols) ? src[(size_t)r * scols + c] : 0.0f;
    dst[idx] = (__bf16)v;
}

// ---------------- per-example front-end: gather, bn0, conv, SE, bn1+relu -> z (bf16) ----------------
__global__ void prep_kernel(const int* __restrict__ eid, const int* __restrict__ rid,
                            const float* __restrict__ ent_emb, const float* __restrict__ rel_emb,
                            const float* __restrict__ rel_transfer, const float* __restrict__ filt_emb,
                            const float* __restrict__ bn0g, const float* __restrict__ bn0b,
                            const float* __restrict__ bn0m, const float* __restrict__ bn0v,
                            const float* __restrict__ seW1, const float* __restrict__ seb1,
                            const float* __restrict__ seW2, const float* __restrict__ seb2,
                            const float* __restrict__ bn1g, const float* __restrict__ bn1b,
                            const float* __restrict__ bn1m, const float* __restrict__ bn1v,
                            __bf16* __restrict__ z) {
    __shared__ float s_e[ENT_DIM];
    __shared__ float s_r[ENT_DIM];
    __shared__ float s_conv[FCK];         // 25.5 KB (well inside 320 KB/WGP)
    __shared__ float s_f[OUT_CH * 4];
    __shared__ float s_y[OUT_CH];
    __shared__ float s_h[8];
    __shared__ float s_A[OUT_CH];
    __shared__ float s_C[OUT_CH];

    const int b = blockIdx.x;
    const int t = threadIdx.x;
    const int ei = eid[b];
    const int ri = rid[b];

    const float sc0 = bn0g[0] * rsqrtf(bn0v[0] + EPS);
    const float m0  = bn0m[0];
    const float bb0 = bn0b[0];

    if (t < ENT_DIM) {
        float he = ent_emb[(size_t)ei * ENT_DIM + t];
        float hr = rel_emb[(size_t)ri * ENT_DIM + t];
        float ht = rel_transfer[(size_t)ri * ENT_DIM + t];
        float ev = he * ht;
        float rv = hr * ev;
        s_e[t] = (ev - m0) * sc0 + bb0;
        s_r[t] = (rv - m0) * sc0 + bb0;
    }
    if (t < OUT_CH * 4) s_f[t] = filt_emb[(size_t)ri * (OUT_CH * 4) + t];
    __syncthreads();

    // 2x2 VALID conv on [e';r'] (2x200) -> (32,199)
    for (int idx = t; idx < FCK; idx += 256) {
        int o = idx / CW;
        int w = idx - o * CW;
        const float* f = &s_f[o * 4];
        s_conv[idx] = f[0] * s_e[w] + f[1] * s_e[w + 1] + f[2] * s_r[w] + f[3] * s_r[w + 1];
    }
    __syncthreads();

    if (t < OUT_CH) {                       // mean over width
        float acc = 0.0f;
        for (int w = 0; w < CW; ++w) acc += s_conv[t * CW + w];
        s_y[t] = acc * (1.0f / (float)CW);
    }
    __syncthreads();

    if (t < 8) {                            // SE: 32 -> 8, relu
        float acc = seb1[t];
        for (int o = 0; o < OUT_CH; ++o) acc += s_y[o] * seW1[t * OUT_CH + o];
        s_h[t] = fmaxf(acc, 0.0f);
    }
    __syncthreads();

    if (t < OUT_CH) {                       // SE: 8 -> 32, sigmoid; fold with bn1 affine
        float acc = seb2[t];
        for (int j = 0; j < 8; ++j) acc += s_h[j] * seW2[t * 8 + j];
        float s   = 1.0f / (1.0f + __expf(-acc));
        float sc1 = bn1g[t] * rsqrtf(bn1v[t] + EPS);
        s_A[t] = s * sc1;                   // z = relu(conv*s*sc1 + (b1 - m1*sc1))
        s_C[t] = bn1b[t] - bn1m[t] * sc1;
    }
    __syncthreads();

    __bf16* zb = z + (size_t)b * FCK;
    for (int idx = t; idx < FCK; idx += 256) {
        int o = idx / CW;
        float zv = fmaxf(s_conv[idx] * s_A[o] + s_C[o], 0.0f);
        zb[idx] = (__bf16)zv;
    }
}

// ---------------- WMMA helper: bf16 A-fragment load per ISA 16x32 layout ----------------
__device__ __forceinline__ v16bf load_a_frag(const __bf16* p) {
    // p points at row (m0 + lane&15), element (k0 + (lane>>4)*8)
    v16bf a;
    *((v8bf*)&a)     = *(const v8bf*)(p);        // K: {0..7}|{8..15}
    *((v8bf*)&a + 1) = *(const v8bf*)(p + 16);   // K: {16..23}|{24..31}
    return a;
}

// ---------------- fc GEMM: xfc = bn2(z @ fcW^T + fc_b), output bf16 [1024 x 224] ----------------
__global__ void fc_gemm_kernel(const __bf16* __restrict__ z, const __bf16* __restrict__ fcw,
                               const float* __restrict__ fcb,
                               const float* __restrict__ g2, const float* __restrict__ b2,
                               const float* __restrict__ m2, const float* __restrict__ v2,
                               __bf16* __restrict__ xfc) {
    const int tid  = threadIdx.x;
    const int wave = tid >> 5;
    const int lane = tid & 31;
    const int lo   = lane & 15;
    const int hi   = lane >> 4;

    const int tile  = blockIdx.x * 8 + wave;     // 0..895  (64 M-tiles x 14 N-tiles)
    const int ntile = tile % 14;
    const int mtile = tile / 14;
    const int m0 = mtile * 16;
    const int n0 = ntile * 16;

    const __bf16* pa = z   + (size_t)(m0 + lo) * FCK + hi * 8;
    const __bf16* pb = fcw + (size_t)(n0 + lo) * FCK + hi * 8;

    v8f acc = {};
    #pragma unroll 4
    for (int k = 0; k < FCK; k += 32) {
        v16bf a = load_a_frag(pa + k);
        v16bf bv = load_a_frag(pb + k);          // same K layout as A
        acc = __builtin_amdgcn_wmma_f32_16x16x32_bf16(false, a, false, bv,
                                                      (short)0, acc, false, false);
    }

    const int d = n0 + lo;
    const bool valid = d < ENT_DIM;
    float fb = 0.0f, sg = 0.0f, bb = 0.0f, mm = 0.0f;
    if (valid) {
        fb = fcb[d];
        sg = g2[d] * rsqrtf(v2[d] + EPS);
        bb = b2[d];
        mm = m2[d];
    }
    #pragma unroll
    for (int v = 0; v < 8; ++v) {
        int m = m0 + v + hi * 8;
        float x = (acc[v] + fb - mm) * sg + bb;
        xfc[(size_t)m * KPAD + d] = (__bf16)(valid ? x : 0.0f);
    }
}

// ---------------- main GEMM: out = sigmoid(xfc @ ent_bf16^T + bias), 1024 x 100000 ----------------
// Register blocking: each wave computes 2 M-tiles x 4 N-tiles (32x64 output).
// Per-tile operand traffic drops to (1/2 + 1/4)*7168B; every B fragment feeds
// two back-to-back WMMAs.
__global__ void logits_kernel(const __bf16* __restrict__ xfc, const __bf16* __restrict__ entb,
                              const float* __restrict__ bias, float* __restrict__ out) {
    const int tid  = threadIdx.x;
    const int wave = tid >> 5;
    const int lane = tid & 31;
    const int lo   = lane & 15;
    const int hi   = lane >> 4;

    const int m0    = blockIdx.y * 32;               // 32 M-blocks of 2 tiles
    const int nbase = blockIdx.x * 512 + wave * 64;  // 8 waves x 4 N-tiles of 16

    const __bf16* pa0 = xfc  + (size_t)(m0 + lo) * KPAD + hi * 8;
    const __bf16* pa1 = pa0 + (size_t)16 * KPAD;
    const __bf16* pb0 = entb + (size_t)(nbase + lo) * KPAD + hi * 8;

    v8f acc[2][4] = {};
    #pragma unroll
    for (int k = 0; k < KPAD; k += 32) {
        v16bf a0 = load_a_frag(pa0 + k);             // reused across 4 N-tiles
        v16bf a1 = load_a_frag(pa1 + k);
        #pragma unroll
        for (int t = 0; t < 4; ++t) {
            v16bf bv = load_a_frag(pb0 + (size_t)t * 16 * KPAD + k);
            acc[0][t] = __builtin_amdgcn_wmma_f32_16x16x32_bf16(false, a0, false, bv,
                                                                (short)0, acc[0][t], false, false);
            acc[1][t] = __builtin_amdgcn_wmma_f32_16x16x32_bf16(false, a1, false, bv,
                                                                (short)0, acc[1][t], false, false);
        }
    }

    #pragma unroll
    for (int t = 0; t < 4; ++t) {
        int n = nbase + t * 16 + lo;
        if (n < NENT) {
            float bs = bias[n];
            #pragma unroll
            for (int mi = 0; mi < 2; ++mi) {
                #pragma unroll
                for (int v = 0; v < 8; ++v) {
                    int m = m0 + mi * 16 + v + hi * 8;
                    float x = acc[mi][t][v] + bs;
                    out[(size_t)m * NENT + n] = 1.0f / (1.0f + __expf(-x));
                }
            }
        }
    }
}

// ---------------- launcher ----------------
extern "C" void kernel_launch(void* const* d_in, const int* in_sizes, int n_in,
                              void* d_out, int out_size, void* d_ws, size_t ws_size,
                              hipStream_t stream) {
    (void)in_sizes; (void)n_in; (void)out_size; (void)ws_size; // needs >= ~61.4 MB ws

    const int*   entity_id    = (const int*)d_in[0];
    const int*   relation_id  = (const int*)d_in[1];
    const float* ent_emb      = (const float*)d_in[2];
    const float* rel_emb      = (const float*)d_in[3];
    const float* rel_transfer = (const float*)d_in[4];
    const float* filt_emb     = (const float*)d_in[5];
    const float* bn0g = (const float*)d_in[6];
    const float* bn0b = (const float*)d_in[7];
    const float* bn0m = (const float*)d_in[8];
    const float* bn0v = (const float*)d_in[9];
    const float* seW1 = (const float*)d_in[10];
    const float* seb1 = (const float*)d_in[11];
    const float* seW2 = (const float*)d_in[12];
    const float* seb2 = (const float*)d_in[13];
    const float* bn1g = (const float*)d_in[14];
    const float* bn1b = (const float*)d_in[15];
    const float* bn1m = (const float*)d_in[16];
    const float* bn1v = (const float*)d_in[17];
    const float* fc_W = (const float*)d_in[18];
    const float* fc_b = (const float*)d_in[19];
    const float* bn2g = (const float*)d_in[20];
    const float* bn2b = (const float*)d_in[21];
    const float* bn2m = (const float*)d_in[22];
    const float* bn2v = (const float*)d_in[23];
    const float* bias = (const float*)d_in[24];

    char* ws = (char*)d_ws;
    __bf16* entb = (__bf16*)(ws + ENT_BF_OFF);
    __bf16* fcwb = (__bf16*)(ws + FCW_BF_OFF);
    __bf16* zbuf = (__bf16*)(ws + Z_OFF);
    __bf16* xfc  = (__bf16*)(ws + XFC_OFF);

    // 1) ent_emb -> bf16, zero-padded [100352 x 224]
    {
        size_t n = (size_t)NE_PAD * KPAD;
        int blocks = (int)((n + 255) / 256);
        cvt_pad_kernel<<<blocks, 256, 0, stream>>>(ent_emb, entb, NENT, ENT_DIM, NE_PAD, KPAD);
    }
    // 2) fc_W -> bf16, zero-padded rows [224 x 6368]
    {
        size_t n = (size_t)FCN_PAD * FCK;
        int blocks = (int)((n + 255) / 256);
        cvt_pad_kernel<<<blocks, 256, 0, stream>>>(fc_W, fcwb, ENT_DIM, FCK, FCN_PAD, FCK);
    }
    // 3) per-example front-end -> z bf16 [1024 x 6368]
    prep_kernel<<<BATCH, 256, 0, stream>>>(entity_id, relation_id, ent_emb, rel_emb,
                                           rel_transfer, filt_emb,
                                           bn0g, bn0b, bn0m, bn0v,
                                           seW1, seb1, seW2, seb2,
                                           bn1g, bn1b, bn1m, bn1v, zbuf);
    // 4) fc GEMM + bn2 -> xfc bf16 [1024 x 224]
    fc_gemm_kernel<<<112, 256, 0, stream>>>(zbuf, fcwb, fc_b, bn2g, bn2b, bn2m, bn2v, xfc);
    // 5) logits GEMM + bias + sigmoid -> d_out [1024 x 100000] f32
    logits_kernel<<<dim3(196, 32), 256, 0, stream>>>(xfc, entb, bias, (float*)d_out);
}
